// encoder_lstm_60833916781136
// MI455X (gfx1250) — matrix operand order repaired and checked by reading, not compiled, not run on hardware
//
#include <hip/hip_runtime.h>
#include <hip/hip_bf16.h>

// ---------------------------------------------------------------------------
// Problem constants (from reference): B=256 T=64 SEQ=512 D=1024 H=512 NL=2 NH=4
// ---------------------------------------------------------------------------
#define B_   256
#define T_   64
#define SEQ_ 512
#define D_   1024
#define H_   512

typedef __bf16 bhalf;
typedef __attribute__((ext_vector_type(16))) bhalf v16bf;
typedef __attribute__((ext_vector_type(8)))  bhalf v8bf;
typedef __attribute__((ext_vector_type(8)))  float v8f;
typedef __attribute__((ext_vector_type(4)))  float v4f;
typedef __attribute__((ext_vector_type(4)))  unsigned int u32x4;
typedef __attribute__((ext_vector_type(8)))  int i32x8;
typedef __attribute__((ext_vector_type(4)))  int i32x4;

__device__ __forceinline__ bhalf f2bf(float f) {
    unsigned u = __builtin_bit_cast(unsigned, f);
    u += 0x7FFFu + ((u >> 16) & 1u);                 // round-to-nearest-even
    unsigned short s = (unsigned short)(u >> 16);
    return __builtin_bit_cast(bhalf, s);
}
__device__ __forceinline__ float sigm(float x) { return 1.0f / (1.0f + __expf(-x)); }

// ---------------------------------------------------------------------------
// Tensor Data Mover: issue a 2D tile load (rows x 32 bf16) global -> LDS.
// D# group0: count=1 | lds_addr | global_addr[56:0] | type=2.
// D# group1: data_size=2B, pad_enable=1, pad_interval=3 (16 DW) and
//            pad_amount=3 (4 DW) -> reproduces the padded 40-half LDS rows.
// Toolchains differ in builtin arity: clang-22 (ROCm) = 5 args, clang-23 = 6.
// ---------------------------------------------------------------------------
#if defined(__clang_major__) && (__clang_major__ >= 23)
#define TDM_CALL(g0, g1, g2, g3) \
    __builtin_amdgcn_tensor_load_to_lds((g0), (g1), (g2), (g3), (i32x8){0,0,0,0,0,0,0,0}, 0)
#else
#define TDM_CALL(g0, g1, g2, g3) \
    __builtin_amdgcn_tensor_load_to_lds((g0), (g1), (g2), (g3), 0)
#endif

__device__ __forceinline__ void tdm_load_tile(unsigned ldsOff, const void* gptr,
                                              int tensorD0, int tensorD1,
                                              int tileD0, int tileD1,
                                              long long strideD0)
{
    const unsigned long long ga = (unsigned long long)gptr;
    u32x4 g0;
    g0[0] = 1u;                                                   // count=1 (valid)
    g0[1] = ldsOff;                                               // lds_addr (bytes)
    g0[2] = (unsigned)(ga & 0xFFFFFFFFull);                       // global_addr[31:0]
    g0[3] = (unsigned)((ga >> 32) & 0x01FFFFFFull) | (2u << 30);  // addr[56:32] | type=2

    i32x8 g1;
    g1[0] = (1 << 16) | (1 << 20) | (3 << 22) | (3 << 25);        // 2B, pad 16DW+4DW
    g1[1] = (int)(((unsigned)tensorD0 & 0xFFFFu) << 16);          // tensor_dim0[15:0]
    g1[2] = (int)(((unsigned)tensorD0 >> 16) |
                  (((unsigned)tensorD1 & 0xFFFFu) << 16));        // dim0 hi | dim1 lo
    g1[3] = (int)(((unsigned)tensorD1 >> 16) |
                  ((unsigned)tileD0 << 16));                      // dim1 hi | tile_dim0
    g1[4] = (int)((unsigned)tileD1 & 0xFFFFu);                    // tile_dim1, tile_dim2=0
    g1[5] = (int)(unsigned)(strideD0 & 0xFFFFFFFFll);             // dim0_stride[31:0]
    g1[6] = (int)(unsigned)((strideD0 >> 32) & 0xFFFFll);         // dim0_stride[47:32]
    g1[7] = 0;

    const i32x4 gz = {0, 0, 0, 0};
    TDM_CALL(g0, g1, gz, gz);
}

// ---------------------------------------------------------------------------
// Core GEMM: C[M,N] = act( A[M,K] @ Bw[N,K]^T + bias[N] + add[M,N] )
//   A, Bw bf16 row-major; out f32 and/or bf16 (either may be null).
//   Block: 256 threads = 8 waves (2x4), block tile 64(M) x 128(N),
//   wave tile 32x32 (4 x v_wmma_f32_16x16x32_bf16 accumulators).
//   A/B staged through LDS, padded stride 40 halves -> conflict-free ds_b128.
//   USE_TDM: wave 0 drives the Tensor Data Mover instead of cooperative loads.
// Requires M%64==0, N%128==0, K%32==0 (true for all uses here).
// ---------------------------------------------------------------------------
template <bool USE_TDM>
__global__ __launch_bounds__(256) void gemm_bf16_wmma(
    const bhalf* __restrict__ A, const bhalf* __restrict__ Bw,
    const float* __restrict__ bias, const float* __restrict__ addsrc,
    float* __restrict__ outF, bhalf* __restrict__ outBf,
    int M, int N, int K, int relu)
{
    __shared__ bhalf lA[64 * 40];
    __shared__ bhalf lB[128 * 40];

    const int tid  = threadIdx.x;
    const int lane = tid & 31;
    const int wave = tid >> 5;
    const int wm   = wave >> 2;   // 0..1  (M within block)
    const int wn   = wave & 3;    // 0..3  (N within block)
    const int mBase = blockIdx.y * 64;
    const int nBase = blockIdx.x * 128;

    v8f acc[2][2] = {};

    // cooperative staging assignments (non-TDM path)
    const int aRow = tid >> 2, aSeg = tid & 3;   // 64 rows x 4 segs of 8 halves
    const int bRow = tid >> 1, bSeg = tid & 1;   // 128 rows x 2 segs of 16 halves
    const long long aSrc = (long long)(mBase + aRow) * K + aSeg * 8;
    const long long bSrc = (long long)(nBase + bRow) * K + bSeg * 16;

    const int lrow  = lane & 15;
    const int khalf = lane >> 4;

    for (int kk = 0; kk < K; kk += 32) {
        if (USE_TDM) {
            if (tid < 32) {       // single wave issues both DMA descriptors
                tdm_load_tile((unsigned)(uintptr_t)lA, A + (long long)mBase * K + kk,
                              K, M, 32, 64, K);
                tdm_load_tile((unsigned)(uintptr_t)lB, Bw + (long long)nBase * K + kk,
                              K, N, 32, 128, K);
                __builtin_amdgcn_s_wait_tensorcnt(0);
            }
        } else {
            // stage A tile 64x32 and B tile 128x32 into LDS (16B-aligned b128s)
            *(v8bf*)(&lA[aRow * 40 + aSeg * 8])   = *(const v8bf*)(&A[aSrc + kk]);
            *(v16bf*)(&lB[bRow * 40 + bSeg * 16]) = *(const v16bf*)(&Bw[bSrc + kk]);
            if (kk + 32 < K) {    // prefetch next K-slab (global_prefetch_b8)
                __builtin_prefetch(&A[aSrc + kk + 32], 0, 1);
                __builtin_prefetch(&Bw[bSrc + kk + 32], 0, 1);
            }
        }
        __syncthreads();

        // A fragment: 16-bit 16x32 layout — lane row = m, lanes 0-15 hold
        // K {0..7,16..23}, lanes 16-31 hold K {8..15,24..31}
        v16bf afrag[2];
#pragma unroll
        for (int s = 0; s < 2; ++s) {
            const bhalf* p = &lA[(wm * 32 + s * 16 + lrow) * 40 + 8 * khalf];
            v8bf c0 = *(const v8bf*)p;
            v8bf c1 = *(const v8bf*)(p + 16);
            v16bf t;
#pragma unroll
            for (int i = 0; i < 8; ++i) { t[i] = c0[i]; t[i + 8] = c1[i]; }
            afrag[s] = t;
        }
        // B fragment: lane col = n, same K split as A ({0..7,16..23} / {8..15,24..31})
        v16bf bfrag[2];
#pragma unroll
        for (int t = 0; t < 2; ++t) {
            const bhalf* q = &lB[(wn * 32 + t * 16 + lrow) * 40 + 8 * khalf];
            v8bf d0 = *(const v8bf*)q;
            v8bf d1 = *(const v8bf*)(q + 16);
            v16bf u;
#pragma unroll
            for (int i = 0; i < 8; ++i) { u[i] = d0[i]; u[i + 8] = d1[i]; }
            bfrag[t] = u;
        }

#pragma unroll
        for (int s = 0; s < 2; ++s)
#pragma unroll
            for (int t = 0; t < 2; ++t)
                acc[s][t] = __builtin_amdgcn_wmma_f32_16x16x32_bf16(
                    false, afrag[s], false, bfrag[t], (short)0, acc[s][t], false, false);
        __syncthreads();
    }

    // epilogue: C/D layout — VGPR r: lanes 0-15 M=r, lanes 16-31 M=r+8; N = lane&15
#pragma unroll
    for (int s = 0; s < 2; ++s) {
#pragma unroll
        for (int t = 0; t < 2; ++t) {
            const int gn = nBase + wn * 32 + t * 16 + lrow;
#pragma unroll
            for (int r = 0; r < 8; ++r) {
                const int gm = mBase + wm * 32 + s * 16 + r + 8 * khalf;
                const long long idx = (long long)gm * N + gn;
                float v = acc[s][t][r];
                if (bias)   v += bias[gn];
                if (addsrc) v += addsrc[idx];
                if (relu)   v = v > 0.0f ? v : 0.0f;
                if (outF)   outF[idx] = v;
                if (outBf)  outBf[idx] = f2bf(v);
            }
        }
    }
}

// ---------------------------------------------------------------------------
// seq_states mean over SEQ axis: (B,SEQ,D) -> (B,D), NT streaming loads (512MB)
// ---------------------------------------------------------------------------
__global__ void seq_mean_kernel(const float* __restrict__ seq,
                                float* __restrict__ pooledF, bhalf* __restrict__ pooledBf)
{
    const int i = blockIdx.x * 256 + threadIdx.x;      // B*D threads
    const int b = i >> 10, d = i & (D_ - 1);
    const float* p = seq + (long long)b * SEQ_ * D_ + d;
    float s = 0.0f;
    for (int k = 0; k < SEQ_; ++k)
        s += __builtin_nontemporal_load(p + (long long)k * D_);
    s *= (1.0f / (float)SEQ_);
    pooledF[i]  = s;
    pooledBf[i] = f2bf(s);
}

// ---------------------------------------------------------------------------
// Layer-0 LSTM: h[0] is never written by the scan -> stays 0, so gates depend
// only on scalar x[b,t]. Compute h0_out for ALL t up front (pure elementwise).
// ---------------------------------------------------------------------------
__global__ void lstm_layer0_kernel(const float* __restrict__ x,
                                   const float* __restrict__ Wih0,
                                   const float* __restrict__ bih0,
                                   const float* __restrict__ bhh0,
                                   bhalf* __restrict__ h0out)
{
    const int i = blockIdx.x * 256 + threadIdx.x;      // B*H threads
    const int b = i >> 9, j = i & (H_ - 1);
    const float wi = Wih0[j],          bi = bih0[j] + bhh0[j];
    const float wf = Wih0[H_ + j],     bf = bih0[H_ + j] + bhh0[H_ + j];
    const float wg = Wih0[2 * H_ + j], bg = bih0[2 * H_ + j] + bhh0[2 * H_ + j];
    const float wo = Wih0[3 * H_ + j], bo = bih0[3 * H_ + j] + bhh0[3 * H_ + j];
    float c = 0.0f;
    for (int t = 0; t < T_; ++t) {
        const float xt = x[b * T_ + t];
        const float ig = sigm(xt * wi + bi);
        const float fg = sigm(xt * wf + bf);
        const float gg = tanhf(xt * wg + bg);
        const float og = sigm(xt * wo + bo);
        c = fg * c + ig * gg;
        h0out[((long long)t * B_ + b) * H_ + j] = f2bf(og * tanhf(c));
    }
}

// layer-1 cell update from gate pre-activations G (B x 4H)
__global__ void lstm_cell_kernel(const float* __restrict__ G,
                                 float* __restrict__ c1,
                                 float* __restrict__ candF, bhalf* __restrict__ candBf)
{
    const int i = blockIdx.x * 256 + threadIdx.x;      // B*H threads
    const int b = i >> 9, j = i & (H_ - 1);
    const float* g = G + (long long)b * (4 * H_);
    const float ig = sigm(g[j]);
    const float fg = sigm(g[H_ + j]);
    const float gg = tanhf(g[2 * H_ + j]);
    const float og = sigm(g[3 * H_ + j]);
    const float c  = fg * c1[i] + ig * gg;
    c1[i] = c;
    const float h = og * tanhf(c);
    candF[i] = h; candBf[i] = f2bf(h);
}

// gate blend: P = [attn_out_pre | gate_pre]; upd = g*cand + (1-g)*attn
__global__ void gate_update_kernel(const float* __restrict__ P,
                                   const float* __restrict__ candF,
                                   float* __restrict__ h1F, bhalf* __restrict__ h1Bf)
{
    const int i = blockIdx.x * 256 + threadIdx.x;      // B*H threads
    const int b = i >> 9, j = i & (H_ - 1);
    const float attn = P[(long long)b * (2 * H_) + j];
    const float gate = sigm(P[(long long)b * (2 * H_) + H_ + j]);
    const float u = gate * candF[i] + (1.0f - gate) * attn;
    h1F[i] = u; h1Bf[i] = f2bf(u);
}

// ---------------------------------------------------------------------------
// One-time prep helpers (tiny f32 work, launched once per call)
// ---------------------------------------------------------------------------
__global__ void prep_gemm_nn(const float* __restrict__ A, int lda,
                             const float* __restrict__ Bm, int ldb,
                             float* __restrict__ C, int N, int K)
{
    const int i = blockIdx.x * 256 + threadIdx.x;
    const int m = i / N, n = i % N;
    float s = 0.0f;
    for (int k = 0; k < K; ++k) s += A[(long long)m * lda + k] * Bm[(long long)k * ldb + n];
    C[i] = s;
}
__global__ void matvec_kernel(const float* __restrict__ A, int lda,
                              const float* __restrict__ x, const float* __restrict__ badd,
                              float* __restrict__ y, int K)
{
    const int m = blockIdx.x * 256 + threadIdx.x;
    float s = badd ? badd[m] : 0.0f;
    for (int k = 0; k < K; ++k) s += A[(long long)m * lda + k] * x[k];
    y[m] = s;
}
__global__ void cast_f32_bf16(const float* __restrict__ s, bhalf* __restrict__ d, int n)
{
    const int i = blockIdx.x * 256 + threadIdx.x;
    if (i < n) d[i] = f2bf(s[i]);
}
__global__ void vec_add_kernel(const float* __restrict__ a, const float* __restrict__ b,
                               float* __restrict__ o, int n)
{
    const int i = blockIdx.x * 256 + threadIdx.x;
    if (i < n) o[i] = a[i] + b[i];
}
// Wcomb = [ Wc ; gW1 + X ] in bf16 (gW1 = gate_W[:, :H], row stride 2H)
__global__ void assemble_wcomb(const float* __restrict__ Wc, const float* __restrict__ X,
                               const float* __restrict__ gateW, bhalf* __restrict__ Wcomb)
{
    const int i = blockIdx.x * 256 + threadIdx.x;      // H*H threads
    const int m = i >> 9, k = i & (H_ - 1);
    Wcomb[i] = f2bf(Wc[i]);
    Wcomb[H_ * H_ + i] = f2bf(gateW[(long long)m * (2 * H_) + k] + X[i]);
}
__global__ void zero_words(unsigned* __restrict__ p, int n)
{
    const int i = blockIdx.x * 256 + threadIdx.x;
    if (i < n) p[i] = 0u;
}

// broadcast (B,H) -> (B,SEQ,H) with NT stores (streams 268MB per call)
__global__ void broadcast_kernel(const float* __restrict__ src, float* __restrict__ dst)
{
    const long long i4 = ((long long)blockIdx.x * 256 + threadIdx.x) * 4;
    const int b = (int)(i4 / ((long long)SEQ_ * H_));
    const int h = (int)(i4 & (H_ - 1));
    const v4f v = *(const v4f*)(src + (long long)b * H_ + h);
    __builtin_nontemporal_store(v, (v4f*)(dst + i4));
}

// ---------------------------------------------------------------------------
extern "C" void kernel_launch(void* const* d_in, const int* in_sizes, int n_in,
                              void* d_out, int out_size, void* d_ws, size_t ws_size,
                              hipStream_t stream)
{
    (void)in_sizes; (void)n_in; (void)out_size; (void)ws_size;
    const float* x        = (const float*)d_in[0];
    const float* seq      = (const float*)d_in[1];
    const float* Wih0     = (const float*)d_in[2];
    const float* bih0     = (const float*)d_in[4];
    const float* bhh0     = (const float*)d_in[5];
    const float* Wih1     = (const float*)d_in[6];
    const float* Whh1     = (const float*)d_in[7];
    const float* bih1     = (const float*)d_in[8];
    const float* bhh1     = (const float*)d_in[9];
    const float* attn_Win = (const float*)d_in[10];
    const float* attn_bin = (const float*)d_in[11];
    const float* attn_Wout= (const float*)d_in[12];
    const float* attn_bout= (const float*)d_in[13];
    const float* gate_W   = (const float*)d_in[16];
    const float* gate_b   = (const float*)d_in[17];
    const float* ovl_W1   = (const float*)d_in[18];
    const float* ovl_b1   = (const float*)d_in[19];
    const float* ovl_W2   = (const float*)d_in[20];
    const float* ovl_b2   = (const float*)d_in[21];
    const float* sap_W    = (const float*)d_in[22];
    const float* sap_b    = (const float*)d_in[23];
    const float* sa_Win   = (const float*)d_in[24];
    const float* sa_bin   = (const float*)d_in[25];
    const float* sa_Wout  = (const float*)d_in[26];
    const float* sa_bout  = (const float*)d_in[27];
    const float* sl_W1    = (const float*)d_in[28];
    const float* sl_b1    = (const float*)d_in[29];
    const float* sl_W2    = (const float*)d_in[30];
    const float* sl_b2    = (const float*)d_in[31];
    const float* sol_W1   = (const float*)d_in[32];
    const float* sol_b1   = (const float*)d_in[33];
    const float* sol_W2   = (const float*)d_in[34];
    const float* sol_b2   = (const float*)d_in[35];
    float* dout = (float*)d_out;

    // ---- workspace carve-out -------------------------------------------------
    char* wsb = (char*)d_ws;
    size_t cur = 0;
    auto alloc = [&](size_t bytes) { size_t o = cur; cur = (cur + bytes + 255) & ~(size_t)255; return (void*)(wsb + o); };

    float* pooledF   = (float*)alloc((size_t)B_ * D_ * 4);
    bhalf* pooledBf  = (bhalf*)alloc((size_t)B_ * D_ * 2);
    bhalf* h0out     = (bhalf*)alloc((size_t)T_ * B_ * H_ * 2);
    float* g1ih      = (float*)alloc((size_t)T_ * B_ * 4 * H_ * 4);   // 134 MB
    float* G         = (float*)alloc((size_t)B_ * 4 * H_ * 4);
    float* P         = (float*)alloc((size_t)B_ * 2 * H_ * 4);
    float* h1F       = (float*)alloc((size_t)B_ * H_ * 4);
    bhalf* h1Bf      = (bhalf*)alloc((size_t)B_ * H_ * 2);
    float* candF     = (float*)alloc((size_t)B_ * H_ * 4);
    bhalf* candBf    = (bhalf*)alloc((size_t)B_ * H_ * 2);
    float* c1        = (float*)alloc((size_t)B_ * H_ * 4);
    bhalf* Wih1bf    = (bhalf*)alloc((size_t)4 * H_ * H_ * 2);
    bhalf* Whh1bf    = (bhalf*)alloc((size_t)4 * H_ * H_ * 2);
    float* b12       = (float*)alloc((size_t)4 * H_ * 4);
    float* WcF       = (float*)alloc((size_t)H_ * H_ * 4);
    float* XF        = (float*)alloc((size_t)H_ * H_ * 4);
    bhalf* Wcomb     = (bhalf*)alloc((size_t)2 * H_ * H_ * 2);
    float* bcomb     = (float*)alloc((size_t)2 * H_ * 4);
    bhalf* ovlW1bf   = (bhalf*)alloc((size_t)H_ * H_ * 2);
    bhalf* ovlW2bf   = (bhalf*)alloc((size_t)H_ * H_ * 2);
    bhalf* sapWbf    = (bhalf*)alloc((size_t)H_ * D_ * 2);
    float* WscF      = (float*)alloc((size_t)H_ * H_ * 4);
    bhalf* Wscbf     = (bhalf*)alloc((size_t)H_ * H_ * 2);
    float* bsc       = (float*)alloc((size_t)H_ * 4);
    bhalf* slW1bf    = (bhalf*)alloc((size_t)H_ * H_ * 2);
    bhalf* slW2bf    = (bhalf*)alloc((size_t)H_ * H_ * 2);
    bhalf* solW1bf   = (bhalf*)alloc((size_t)H_ * H_ * 2);
    bhalf* solW2bf   = (bhalf*)alloc((size_t)H_ * H_ * 2);
    bhalf* t1bf      = (bhalf*)alloc((size_t)B_ * H_ * 2);
    float* ovF       = (float*)alloc((size_t)B_ * H_ * 4);
    float* sspF      = (float*)alloc((size_t)B_ * H_ * 4);
    bhalf* sspBf     = (bhalf*)alloc((size_t)B_ * H_ * 2);
    bhalf* zBf       = (bhalf*)alloc((size_t)B_ * H_ * 2);
    bhalf* t2bf      = (bhalf*)alloc((size_t)B_ * H_ * 2);
    bhalf* seqoutBf  = (bhalf*)alloc((size_t)B_ * H_ * 2);
    bhalf* t3bf      = (bhalf*)alloc((size_t)B_ * H_ * 2);
    float* soF       = (float*)alloc((size_t)B_ * H_ * 4);

    const dim3 blk(256);
    auto gemm = [&](const bhalf* A, const bhalf* Bw, const float* bias, const float* add,
                    float* oF, bhalf* oBf, int M, int N, int K, int relu) {
        gemm_bf16_wmma<false><<<dim3(N / 128, M / 64), blk, 0, stream>>>(A, Bw, bias, add, oF, oBf, M, N, K, relu);
    };
    auto gemm_tdm = [&](const bhalf* A, const bhalf* Bw, const float* bias, const float* add,
                        float* oF, bhalf* oBf, int M, int N, int K, int relu) {
        gemm_bf16_wmma<true><<<dim3(N / 128, M / 64), blk, 0, stream>>>(A, Bw, bias, add, oF, oBf, M, N, K, relu);
    };
    auto ew = [&](int n) { return dim3((unsigned)((n + 255) / 256)); };

    // ---- init recurrent state to zero ---------------------------------------
    zero_words<<<ew(B_ * H_), blk, 0, stream>>>((unsigned*)h1F, B_ * H_);
    zero_words<<<ew(B_ * H_ / 2), blk, 0, stream>>>((unsigned*)h1Bf, B_ * H_ / 2);
    zero_words<<<ew(B_ * H_), blk, 0, stream>>>((unsigned*)c1, B_ * H_);

    // ---- streaming mean + layer-0 precompute --------------------------------
    seq_mean_kernel<<<ew(B_ * D_), blk, 0, stream>>>(seq, pooledF, pooledBf);
    lstm_layer0_kernel<<<ew(B_ * H_), blk, 0, stream>>>(x, Wih0, bih0, bhh0, h0out);

    // ---- weight prep: bf16 casts + attention/gate folding -------------------
    cast_f32_bf16<<<ew(4 * H_ * H_), blk, 0, stream>>>(Wih1, Wih1bf, 4 * H_ * H_);
    cast_f32_bf16<<<ew(4 * H_ * H_), blk, 0, stream>>>(Whh1, Whh1bf, 4 * H_ * H_);
    cast_f32_bf16<<<ew(H_ * H_), blk, 0, stream>>>(ovl_W1, ovlW1bf, H_ * H_);
    cast_f32_bf16<<<ew(H_ * H_), blk, 0, stream>>>(ovl_W2, ovlW2bf, H_ * H_);
    cast_f32_bf16<<<ew(H_ * D_), blk, 0, stream>>>(sap_W, sapWbf, H_ * D_);
    cast_f32_bf16<<<ew(H_ * H_), blk, 0, stream>>>(sl_W1, slW1bf, H_ * H_);
    cast_f32_bf16<<<ew(H_ * H_), blk, 0, stream>>>(sl_W2, slW2bf, H_ * H_);
    cast_f32_bf16<<<ew(H_ * H_), blk, 0, stream>>>(sol_W1, solW1bf, H_ * H_);
    cast_f32_bf16<<<ew(H_ * H_), blk, 0, stream>>>(sol_W2, solW2bf, H_ * H_);
    vec_add_kernel<<<ew(4 * H_), blk, 0, stream>>>(bih1, bhh1, b12, 4 * H_);

    // key-length-1 attention collapses: Wc = Wout @ Wv, bc = Wout@bv + bout
    const float* attnWv = attn_Win + (size_t)2 * H_ * H_;   // rows [2H,3H)
    const float* attnbv = attn_bin + 2 * H_;
    prep_gemm_nn<<<ew(H_ * H_), blk, 0, stream>>>(attn_Wout, H_, attnWv, H_, WcF, H_, H_);
    prep_gemm_nn<<<ew(H_ * H_), blk, 0, stream>>>(gate_W + H_, 2 * H_, WcF, H_, XF, H_, H_); // gW2 @ Wc
    assemble_wcomb<<<ew(H_ * H_), blk, 0, stream>>>(WcF, XF, gate_W, Wcomb);
    matvec_kernel<<<ew(H_), blk, 0, stream>>>(attn_Wout, H_, attnbv, attn_bout, bcomb, H_);          // bc
    matvec_kernel<<<ew(H_), blk, 0, stream>>>(gate_W + H_, 2 * H_, bcomb, gate_b, bcomb + H_, H_);   // beff

    // seq-attention fold: Wsc = saWout @ saWv, bsc = saWout@sabv + sabout
    const float* saWv = sa_Win + (size_t)2 * H_ * H_;
    const float* sabv = sa_bin + 2 * H_;
    prep_gemm_nn<<<ew(H_ * H_), blk, 0, stream>>>(sa_Wout, H_, saWv, H_, WscF, H_, H_);
    cast_f32_bf16<<<ew(H_ * H_), blk, 0, stream>>>(WscF, Wscbf, H_ * H_);
    matvec_kernel<<<ew(H_), blk, 0, stream>>>(sa_Wout, H_, sabv, sa_bout, bsc, H_);

    // ---- big parallel pre-GEMM (TDM-staged): g1_ih = h0out @ Wih1^T + b12 ----
    gemm_tdm(h0out, Wih1bf, b12, nullptr, g1ih, nullptr, T_ * B_, 4 * H_, H_, 0);

    // ---- sequential scan: 64 steps, 4 kernels each (latency path: coop loads)
    for (int t = 0; t < T_; ++t) {
        gemm(h1Bf, Whh1bf, nullptr, g1ih + (size_t)t * B_ * 4 * H_, G, nullptr, B_, 4 * H_, H_, 0);
        lstm_cell_kernel<<<ew(B_ * H_), blk, 0, stream>>>(G, c1, candF, candBf);
        gemm(candBf, Wcomb, bcomb, nullptr, P, nullptr, B_, 2 * H_, H_, 0);
        gate_update_kernel<<<ew(B_ * H_), blk, 0, stream>>>(P, candF, h1F, h1Bf);
    }

    // ---- heads: ovl MLP on final h1 -----------------------------------------
    gemm(h1Bf, ovlW1bf, ovl_b1, nullptr, nullptr, t1bf, B_, H_, H_, 1);
    gemm(t1bf, ovlW2bf, ovl_b2, nullptr, ovF, nullptr, B_, H_, H_, 0);

    // ---- seq branch ---------------------------------------------------------
    gemm(pooledBf, sapWbf, sap_b, nullptr, sspF, sspBf, B_, H_, D_, 0);        // ssp
    gemm(sspBf, Wscbf, bsc, sspF, nullptr, zBf, B_, H_, H_, 0);                // ssp + attn(ssp)
    gemm(zBf, slW1bf, sl_b1, nullptr, nullptr, t2bf, B_, H_, H_, 1);
    gemm(t2bf, slW2bf, sl_b2, nullptr, nullptr, seqoutBf, B_, H_, H_, 0);
    gemm(seqoutBf, solW1bf, sol_b1, nullptr, nullptr, t3bf, B_, H_, H_, 1);
    gemm(t3bf, solW2bf, sol_b2, nullptr, soF, nullptr, B_, H_, H_, 0);

    // ---- broadcast writes (536 MB, NT stores) -------------------------------
    const long long half = (long long)B_ * SEQ_ * H_;
    broadcast_kernel<<<dim3((unsigned)(half / 4 / 256)), blk, 0, stream>>>(ovF, dout);
    broadcast_kernel<<<dim3((unsigned)(half / 4 / 256)), blk, 0, stream>>>(soF, dout + half);
}
